// TensorNet_58531814310163
// MI455X (gfx1250) — matrix operand order repaired and checked by reading, not compiled, not run on hardware
//
#include <hip/hip_runtime.h>
#include <math.h>

// ---------------- problem constants ----------------
#define NN  10000
#define EE  120000
#define CCH 64
#define RR  32
#define NCF ((size_t)NN * CCH)   // one compact plane: N*C floats
#define PIF 3.14159265358979323846f

typedef _Float16 h16;
typedef __attribute__((ext_vector_type(16))) _Float16 v16h;
typedef __attribute__((ext_vector_type(8)))  float    v8f;

__device__ __forceinline__ float siluf(float x) { return x / (1.f + expf(-x)); }
__device__ __forceinline__ void atomAdd(float* p, float v) {
  __hip_atomic_fetch_add(p, v, __ATOMIC_RELAXED, __HIP_MEMORY_SCOPE_AGENT);
}

// =====================================================================
// Generic WMMA GEMM:  out[M x Nc] = epilogue( A[M x K](f16) @ W[Nc x K]^T + b )
// One wave per 16x16 tile, K stepped by 32 via v_wmma_f32_16x16x32_f16.
// epilogue: optional SiLU, optional per-row scale (cosine cutoff), f16 or f32 out.
// =====================================================================
__global__ void wmma_gemm_kernel(const h16* __restrict__ A, const h16* __restrict__ W,
                                 const float* __restrict__ bias, const float* __restrict__ cc,
                                 h16* __restrict__ outH, float* __restrict__ outF,
                                 int M, int K, int Nc, int act, int mulcc)
{
  const int lane = threadIdx.x & 31;
  const int wave = (int)((blockIdx.x * blockDim.x + threadIdx.x) >> 5);
  const int nTiles = Nc >> 4;
  const int tiles  = (M >> 4) * nTiles;
  if (wave >= tiles) return;                       // uniform per-wave exit; EXEC stays full
  const int mT = wave / nTiles;
  const int nT = wave - mT * nTiles;
  const int l16 = lane & 15;
  const int hi  = lane >> 4;
  const int mRow = (mT << 4) + l16;                // A row this lane feeds
  const int nCol = (nT << 4) + l16;                // B column / D column this lane holds
  v8f acc = {0.f, 0.f, 0.f, 0.f, 0.f, 0.f, 0.f, 0.f};

  for (int kk = 0; kk < K; kk += 32) {
    // A fragment: 16-bit A 16x32 layout. lane<16: K pairs {0,2,4,6,16,18,20,22};
    // lane>=16: +8. arow pre-offset by hi*8.
    v16h a;
    const h16* arow = A + (size_t)mRow * K + kk + (hi << 3);
#pragma unroll
    for (int i = 0; i < 4; ++i) {
      a[2*i]     = arow[2*i];
      a[2*i + 1] = arow[2*i + 1];
      a[2*i + 8] = arow[16 + 2*i];
      a[2*i + 9] = arow[16 + 2*i + 1];
    }
    // B fragment: 32x16 (KxN); same K order as A: elems 0-7 -> K hi*8+0..7,
    // elems 8-15 -> K 16+hi*8+0..7.  B[k][n] = W[n][k]  (lin(): y = x @ W^T)
    v16h b;
    const h16* wrow = W + (size_t)nCol * K + kk + (hi << 3);
#pragma unroll
    for (int i = 0; i < 16; ++i) b[i] = wrow[i < 8 ? i : i + 8];

    acc = __builtin_amdgcn_wmma_f32_16x16x32_f16(false, a, false, b,
                                                 (short)0, acc, false, false);
  }

  const float bv = bias ? bias[nCol] : 0.f;
#pragma unroll
  for (int r = 0; r < 8; ++r) {
    const int mOut = (mT << 4) + (hi << 3) + r;    // f32 D layout: VGPR r -> M = r + hi*8
    float v = acc[r] + bv;
    if (act)   v = siluf(v);
    if (mulcc) v *= cc[mOut];
    const size_t o = (size_t)mOut * Nc + nCol;
    if (outH) outH[o] = (h16)v; else outF[o] = v;
  }
}

// =====================================================================
// Elementwise helpers
// =====================================================================
__global__ void f2h_kernel(const float* __restrict__ s, h16* __restrict__ d, int n) {
  int i = blockIdx.x * blockDim.x + threadIdx.x;
  if (i < n) d[i] = (h16)s[i];
}
__global__ void zero_kernel(float* __restrict__ p, int n) {
  int i = blockIdx.x * blockDim.x + threadIdx.x;
  if (i < n) p[i] = 0.f;
}

// RBF + cosine cutoff (edge_attr in f16, Cc in f32)
__global__ void rbf_kernel(const float* __restrict__ ew, h16* __restrict__ eattr,
                           float* __restrict__ cc)
{
  int e = blockIdx.x * blockDim.x + threadIdx.x;
  if (e >= EE) return;
  float d = ew[e];
  float c = (d < 4.5f) ? 0.5f * (cosf(d * PIF / 4.5f) + 1.f) : 0.f;
  cc[e] = c;
  const float start = expf(-4.5f);
  const float bden  = (2.f / 32.f) * (1.f - start);
  const float beta  = 1.f / (bden * bden);
  const float alpha = 5.f / 4.5f;
  float ex = expf(-alpha * d);
#pragma unroll
  for (int r = 0; r < RR; ++r) {
    float mean = start + (1.f - start) * ((float)r / 31.f);
    float t = ex - mean;
    eattr[(size_t)e * RR + r] = (h16)(c * expf(-beta * t * t));
  }
}

// Zcat[e] = [ emb[z[dst]] , emb[z[src]] ]  in f16
__global__ void zcat_kernel(const int* __restrict__ z, const int* __restrict__ ei,
                            const float* __restrict__ emb, h16* __restrict__ zc)
{
  int id = blockIdx.x * blockDim.x + threadIdx.x;
  if (id >= EE * CCH) return;
  int e = id >> 6, c = id & 63;
  int s = ei[e], d = ei[EE + e];
  zc[(size_t)e * 128 + c]      = (h16)emb[(size_t)z[d] * CCH + c];
  zc[(size_t)e * 128 + 64 + c] = (h16)emb[(size_t)z[s] * CCH + c];
}

// Compact 10-plane representation of a 3x3 per (n,c) tensor:
//   plane0: lambda (I part),  planes1-3: p=A01,q=A02,r=A12 (skew),
//   planes4-9: s00,s01,s02,s11,s12,s22 (sym traceless)
__device__ __forceinline__ void load_full(const float* __restrict__ P, size_t off,
                                          float X[3][3])
{
  float l   = P[off];
  float p   = P[NCF + off],   q   = P[2*NCF + off], r   = P[3*NCF + off];
  float s00 = P[4*NCF + off], s01 = P[5*NCF + off], s02 = P[6*NCF + off];
  float s11 = P[7*NCF + off], s12 = P[8*NCF + off], s22 = P[9*NCF + off];
  X[0][0] = l + s00;  X[0][1] =  p + s01; X[0][2] =  q + s02;
  X[1][0] = -p + s01; X[1][1] = l + s11;  X[1][2] =  r + s12;
  X[2][0] = -q + s02; X[2][1] = -r + s12; X[2][2] = l + s22;
}
__device__ __forceinline__ void mm3(const float A[3][3], const float B[3][3],
                                    float C[3][3])
{
#pragma unroll
  for (int i = 0; i < 3; ++i)
#pragma unroll
    for (int j = 0; j < 3; ++j)
      C[i][j] = A[i][0]*B[0][j] + A[i][1]*B[1][j] + A[i][2]*B[2][j];
}

// Embedding scatter: per (edge, channel) accumulate Zij*(W1*I + W2*skew(ev) + W3*sym(ev))
__global__ void emb_scatter_kernel(const int* __restrict__ ei, const float* __restrict__ evec,
                                   const h16* __restrict__ w1, const h16* __restrict__ w2,
                                   const h16* __restrict__ w3, const h16* __restrict__ zij,
                                   float* __restrict__ P)
{
  int id = blockIdx.x * blockDim.x + threadIdx.x;
  if (id >= EE * CCH) return;
  int e = id >> 6, c = id & 63;
  int s = ei[e], d = ei[EE + e];
  float vx = evec[3*(size_t)e], vy = evec[3*(size_t)e + 1], vz = evec[3*(size_t)e + 2];
  if (s != d) {
    float nv = sqrtf(vx*vx + vy*vy + vz*vz);
    float iv = (nv == 0.f) ? 1.f : (1.f / nv);
    vx *= iv; vy *= iv; vz *= iv;
  }
  size_t eo = (size_t)e * CCH + c;
  float zf = (float)zij[eo];
  float W1 = (float)w1[eo] * zf;
  float W2 = (float)w2[eo] * zf;
  float W3 = (float)w3[eo] * zf;
  size_t off = (size_t)d * CCH + c;
  atomAdd(P + off, W1);
  atomAdd(P + NCF   + off, -vz * W2);   // A01 of skew(ev)
  atomAdd(P + 2*NCF + off,  vy * W2);   // A02
  atomAdd(P + 3*NCF + off, -vx * W2);   // A12
  float n3 = (vx*vx + vy*vy + vz*vz) * (1.f / 3.f);
  atomAdd(P + 4*NCF + off, (vx*vx - n3) * W3);
  atomAdd(P + 5*NCF + off, (vx*vy)      * W3);
  atomAdd(P + 6*NCF + off, (vx*vz)      * W3);
  atomAdd(P + 7*NCF + off, (vy*vy - n3) * W3);
  atomAdd(P + 8*NCF + off, (vy*vz)      * W3);
  atomAdd(P + 9*NCF + off, (vz*vz - n3) * W3);
}

// tensor_norm of I+A+S from compact planes, then LayerNorm over C=64 -> f16
__global__ void tn_ln64_kernel(const float* __restrict__ P, const float* __restrict__ lw,
                               const float* __restrict__ lb, h16* __restrict__ out)
{
  int n = blockIdx.x, c = threadIdx.x;
  size_t off = (size_t)n * CCH + c;
  float l = P[off];
  float p = P[NCF+off], q = P[2*NCF+off], r = P[3*NCF+off];
  float s00 = P[4*NCF+off], s01 = P[5*NCF+off], s02 = P[6*NCF+off];
  float s11 = P[7*NCF+off], s12 = P[8*NCF+off], s22 = P[9*NCF+off];
  float t = 3.f*l*l + 2.f*(p*p + q*q + r*r)
          + (s00*s00 + s11*s11 + s22*s22) + 2.f*(s01*s01 + s02*s02 + s12*s12);
  __shared__ float sh[64], sh2[64];
  sh[c] = t; sh2[c] = t * t;
  __syncthreads();
  for (int s2 = 32; s2 > 0; s2 >>= 1) {
    if (c < s2) { sh[c] += sh[c + s2]; sh2[c] += sh2[c + s2]; }
    __syncthreads();
  }
  float mean = sh[0] * (1.f / 64.f);
  float var  = sh2[0] * (1.f / 64.f) - mean * mean;
  out[off] = (h16)((t - mean) * rsqrtf(var + 1e-5f) * lw[c] + lb[c]);
}

// X = n0*I' + n1*A' + n2*S'   (norm scalars from ls2 output, (N,C,3) layout)
__global__ void emb_combine_kernel(const float* __restrict__ Yp, const h16* __restrict__ n3,
                                   float* __restrict__ Xp)
{
  int id = blockIdx.x * blockDim.x + threadIdx.x;
  if (id >= (int)NCF) return;
  int n = id >> 6, c = id & 63;
  size_t off = (size_t)id;
  float n0 = (float)n3[(size_t)n*192 + 3*c];
  float n1 = (float)n3[(size_t)n*192 + 3*c + 1];
  float n2 = (float)n3[(size_t)n*192 + 3*c + 2];
  Xp[off] = n0 * Yp[off];
#pragma unroll
  for (int pl = 1; pl <= 3; ++pl) Xp[pl*NCF + off] = n1 * Yp[pl*NCF + off];
#pragma unroll
  for (int pl = 4; pl <= 9; ++pl) Xp[pl*NCF + off] = n2 * Yp[pl*NCF + off];
}

// X = X / (||X||_F^2 + 1); write f32 in place + f16 copy for WMMA
__global__ void x_normalize_kernel(float* __restrict__ Xp, h16* __restrict__ Xh)
{
  int id = blockIdx.x * blockDim.x + threadIdx.x;
  if (id >= (int)NCF) return;
  size_t off = (size_t)id;
  float l = Xp[off];
  float p = Xp[NCF+off], q = Xp[2*NCF+off], r = Xp[3*NCF+off];
  float s00 = Xp[4*NCF+off], s01 = Xp[5*NCF+off], s02 = Xp[6*NCF+off];
  float s11 = Xp[7*NCF+off], s12 = Xp[8*NCF+off], s22 = Xp[9*NCF+off];
  float tn = 3.f*l*l + 2.f*(p*p + q*q + r*r)
           + (s00*s00 + s11*s11 + s22*s22) + 2.f*(s01*s01 + s02*s02 + s12*s12);
  float sc = 1.f / (tn + 1.f);
#pragma unroll
  for (int pl = 0; pl < 10; ++pl) {
    float v = Xp[pl*NCF + off] * sc;
    Xp[pl*NCF + off] = v;
    Xh[pl*NCF + off] = (h16)v;
  }
}

// Message scatter: msg[dst] += ea_k * Y_plane[src]  (k=0: I, k=1: A, k=2: S)
__global__ void inter_scatter_kernel(const int* __restrict__ ei, const h16* __restrict__ ea,
                                     const float* __restrict__ Yp, float* __restrict__ Mp)
{
  int id = blockIdx.x * blockDim.x + threadIdx.x;
  if (id >= EE * CCH) return;
  int e = id >> 6, c = id & 63;
  int s = ei[e], d = ei[EE + e];
  float e0 = (float)ea[(size_t)e*192 + 3*c];
  float e1 = (float)ea[(size_t)e*192 + 3*c + 1];
  float e2 = (float)ea[(size_t)e*192 + 3*c + 2];
  size_t so = (size_t)s * CCH + c, dof = (size_t)d * CCH + c;
  atomAdd(Mp + dof, e0 * Yp[so]);
#pragma unroll
  for (int pl = 1; pl <= 3; ++pl) atomAdd(Mp + pl*NCF + dof, e1 * Yp[pl*NCF + so]);
#pragma unroll
  for (int pl = 4; pl <= 9; ++pl) atomAdd(Mp + pl*NCF + dof, e2 * Yp[pl*NCF + so]);
}

// P = msg@Y + Y@msg; decompose; scale by 1/(||P||^2+1); emit compact f16
__global__ void inter_combine_kernel(const float* __restrict__ Mp, const float* __restrict__ Yp,
                                     h16* __restrict__ Oh)
{
  int id = blockIdx.x * blockDim.x + threadIdx.x;
  if (id >= (int)NCF) return;
  size_t off = (size_t)id;
  float Mx[3][3], Yx[3][3], P1[3][3], P2[3][3], P[3][3];
  load_full(Mp, off, Mx);
  load_full(Yp, off, Yx);
  mm3(Mx, Yx, P1); mm3(Yx, Mx, P2);
  float fro = 0.f;
#pragma unroll
  for (int i = 0; i < 3; ++i)
#pragma unroll
    for (int j = 0; j < 3; ++j) { P[i][j] = P1[i][j] + P2[i][j]; fro += P[i][j]*P[i][j]; }
  float inv = 1.f / (fro + 1.f);
  float lam = (P[0][0] + P[1][1] + P[2][2]) * (1.f/3.f);
  Oh[off]          = (h16)(lam * inv);
  Oh[NCF   + off]  = (h16)(0.5f*(P[0][1]-P[1][0]) * inv);
  Oh[2*NCF + off]  = (h16)(0.5f*(P[0][2]-P[2][0]) * inv);
  Oh[3*NCF + off]  = (h16)(0.5f*(P[1][2]-P[2][1]) * inv);
  Oh[4*NCF + off]  = (h16)((P[0][0]-lam) * inv);
  Oh[5*NCF + off]  = (h16)(0.5f*(P[0][1]+P[1][0]) * inv);
  Oh[6*NCF + off]  = (h16)(0.5f*(P[0][2]+P[2][0]) * inv);
  Oh[7*NCF + off]  = (h16)((P[1][1]-lam) * inv);
  Oh[8*NCF + off]  = (h16)(0.5f*(P[1][2]+P[2][1]) * inv);
  Oh[9*NCF + off]  = (h16)((P[2][2]-lam) * inv);
}

// dX = dX + dX@dX;  X <- Xnorm + dX  (decomposed back into compact planes, in place)
__global__ void x_update_kernel(const float* __restrict__ Dp, float* __restrict__ Xp)
{
  int id = blockIdx.x * blockDim.x + threadIdx.x;
  if (id >= (int)NCF) return;
  size_t off = (size_t)id;
  float D[3][3], X[3][3], DD[3][3], Z[3][3];
  load_full(Dp, off, D);
  load_full(Xp, off, X);
  mm3(D, D, DD);
#pragma unroll
  for (int i = 0; i < 3; ++i)
#pragma unroll
    for (int j = 0; j < 3; ++j) Z[i][j] = X[i][j] + D[i][j] + DD[i][j];
  float lam = (Z[0][0] + Z[1][1] + Z[2][2]) * (1.f/3.f);
  Xp[off]         = lam;
  Xp[NCF   + off] = 0.5f*(Z[0][1]-Z[1][0]);
  Xp[2*NCF + off] = 0.5f*(Z[0][2]-Z[2][0]);
  Xp[3*NCF + off] = 0.5f*(Z[1][2]-Z[2][1]);
  Xp[4*NCF + off] = Z[0][0]-lam;
  Xp[5*NCF + off] = 0.5f*(Z[0][1]+Z[1][0]);
  Xp[6*NCF + off] = 0.5f*(Z[0][2]+Z[2][0]);
  Xp[7*NCF + off] = Z[1][1]-lam;
  Xp[8*NCF + off] = 0.5f*(Z[1][2]+Z[2][1]);
  Xp[9*NCF + off] = Z[2][2]-lam;
}

// Final: xcat = LN192([tn(I), tn(A), tn(S)]) -> f16 (feeds out_lin WMMA GEMM)
__global__ void final_ln_kernel(const float* __restrict__ Xp, const float* __restrict__ lw,
                                const float* __restrict__ lb, h16* __restrict__ xcat)
{
  int n = blockIdx.x, c = threadIdx.x;     // blockDim = 256, 192 active
  float t = 0.f;
  if (c < 192) {
    int ch = c & 63;
    size_t off = (size_t)n * CCH + ch;
    if (c < 64) {
      float l = Xp[off]; t = 3.f * l * l;
    } else if (c < 128) {
      float p = Xp[NCF+off], q = Xp[2*NCF+off], r = Xp[3*NCF+off];
      t = 2.f * (p*p + q*q + r*r);
    } else {
      float s00 = Xp[4*NCF+off], s01 = Xp[5*NCF+off], s02 = Xp[6*NCF+off];
      float s11 = Xp[7*NCF+off], s12 = Xp[8*NCF+off], s22 = Xp[9*NCF+off];
      t = s00*s00 + s11*s11 + s22*s22 + 2.f*(s01*s01 + s02*s02 + s12*s12);
    }
  }
  __shared__ float sh[256], sh2[256];
  sh[c] = t; sh2[c] = t * t;
  __syncthreads();
  for (int s2 = 128; s2 > 0; s2 >>= 1) {
    if (c < s2) { sh[c] += sh[c + s2]; sh2[c] += sh2[c + s2]; }
    __syncthreads();
  }
  if (c < 192) {
    float mean = sh[0] * (1.f / 192.f);
    float var  = sh2[0] * (1.f / 192.f) - mean * mean;
    xcat[(size_t)n*192 + c] = (h16)((t - mean) * rsqrtf(var + 1e-5f) * lw[c] + lb[c]);
  }
}

// =====================================================================
// Host side
// =====================================================================
// d_in layout: insertion order for setup dict, jax sorted-key order for params:
//  0 z, 1 edge_index, 2 edge_weight, 3 edge_vec,
//  4 dp1_b, 5 dp1_w, 6 dp2_b, 7 dp2_w, 8 dp3_b, 9 dp3_w, 10 emb, 11 emb2_b,
//  12 emb2_w, 13 ln_b, 14 ln_w, 15 ls1_b, 16 ls1_w, 17 ls2_b, 18 ls2_w,
//  19-21 emb.lt[0..2],
//  layer l (l=0,1) base = 22+13*l: +0 l1_b, +1 l1_w, +2 l2_b, +3 l2_w,
//     +4 l3_b, +5 l3_w, +6..+11 lt[0..5],
//  48 out_lin_b, 49 out_lin_w, 50 out_ln_b, 51 out_ln_w
extern "C" void kernel_launch(void* const* d_in, const int* in_sizes, int n_in,
                              void* d_out, int out_size, void* d_ws, size_t ws_size,
                              hipStream_t stream)
{
  (void)in_sizes; (void)n_in; (void)out_size; (void)ws_size;
  const int*   z  = (const int*)d_in[0];
  const int*   ei = (const int*)d_in[1];
  const float* ew = (const float*)d_in[2];
  const float* ev = (const float*)d_in[3];
  auto F = [&](int i) -> const float* { return (const float*)d_in[i]; };

  // bump allocator over workspace
  char* wsbase = (char*)d_ws;
  size_t cur = 0;
  auto alloc = [&](size_t bytes) -> void* {
    void* p = wsbase + cur;
    cur = (cur + bytes + 255) & ~(size_t)255;
    return p;
  };
  auto wcvt = [&](int idx, int n) -> h16* {
    h16* p = (h16*)alloc((size_t)n * sizeof(h16));
    f2h_kernel<<<(n + 255) / 256, 256, 0, stream>>>(F(idx), p, n);
    return p;
  };
  auto gemm = [&](const h16* A, const h16* W, const float* bias, const float* cc,
                  h16* oh, float* of, int M, int K, int Nc, int act, int mulcc) {
    int tiles  = (M >> 4) * (Nc >> 4);
    int blocks = (tiles + 7) >> 3;                 // 8 waves (256 thr) per block
    wmma_gemm_kernel<<<blocks, 256, 0, stream>>>(A, W, bias, cc, oh, of,
                                                 M, K, Nc, act, mulcc);
  };

  // f16 weights
  h16* dp1w = wcvt(5,  CCH * RR);
  h16* dp2w = wcvt(7,  CCH * RR);
  h16* dp3w = wcvt(9,  CCH * RR);
  h16* emb2 = wcvt(12, CCH * 128);
  h16* ls1w = wcvt(16, 128 * CCH);
  h16* ls2w = wcvt(18, 192 * 128);
  h16* elt[3];
  for (int i = 0; i < 3; ++i) elt[i] = wcvt(19 + i, CCH * CCH);
  h16 *l1w[2], *l2w[2], *l3w[2], *llt[2][6];
  int pb[2];
  for (int l = 0; l < 2; ++l) {
    pb[l] = 22 + 13 * l;
    l1w[l] = wcvt(pb[l] + 1, CCH * RR);
    l2w[l] = wcvt(pb[l] + 3, 128 * CCH);
    l3w[l] = wcvt(pb[l] + 5, 192 * 128);
    for (int j = 0; j < 6; ++j) llt[l][j] = wcvt(pb[l] + 6 + j, CCH * CCH);
  }
  h16* outw = wcvt(49, CCH * 192);

  // edge-side buffers
  h16*   eattr = (h16*)alloc((size_t)EE * RR * 2);
  float* ccv   = (float*)alloc((size_t)EE * 4);
  h16*   ebuf  = (h16*)alloc((size_t)EE * 384 * 2);   // shared: embedding temps / ea chain
  h16* zcat = ebuf;                       // E*128
  h16* zij  = ebuf + (size_t)EE * 128;    // E*64
  h16* w1h  = ebuf + (size_t)EE * 192;    // E*64
  h16* w2h  = ebuf + (size_t)EE * 256;    // E*64
  h16* w3h  = ebuf + (size_t)EE * 320;    // E*64
  h16* ea1  = ebuf;                       // E*64   (interaction reuse)
  h16* ea2  = ebuf + (size_t)EE * 64;     // E*128
  h16* ea3  = ebuf + (size_t)EE * 192;    // E*192

  // node-side compact-plane buffers (10 planes of N*C each)
  float* Xp  = (float*)alloc(10 * NCF * 4);
  h16*   Xh  = (h16*)alloc(10 * NCF * 2);
  float* Yp  = (float*)alloc(10 * NCF * 4);
  float* Mp  = (float*)alloc(10 * NCF * 4);
  h16*   Chh = (h16*)alloc(10 * NCF * 2);
  float* Dp  = (float*)alloc(10 * NCF * 4);
  h16* normH = (h16*)alloc(NCF * 2);
  h16* ls1o  = (h16*)alloc((size_t)NN * 128 * 2);
  h16* ls2o  = (h16*)alloc((size_t)NN * 192 * 2);
  h16* xcat  = (h16*)alloc((size_t)NN * 192 * 2);

  const int EC  = EE * CCH;                 // 7.68M
  const int ECB = (EC + 255) / 256;
  const int NCB = ((int)NCF + 255) / 256;
  const int PLN = (int)(10 * NCF);
  const int PLB = (PLN + 255) / 256;

  // ---------- featurization ----------
  rbf_kernel<<<(EE + 255) / 256, 256, 0, stream>>>(ew, eattr, ccv);
  zcat_kernel<<<ECB, 256, 0, stream>>>(z, ei, F(10), zcat);

  // ---------- tensor embedding ----------
  gemm(eattr, dp1w, F(4), ccv, w1h, nullptr, EE, RR, CCH, 0, 1);
  gemm(eattr, dp2w, F(6), ccv, w2h, nullptr, EE, RR, CCH, 0, 1);
  gemm(eattr, dp3w, F(8), ccv, w3h, nullptr, EE, RR, CCH, 0, 1);
  gemm(zcat,  emb2, F(11), nullptr, zij, nullptr, EE, 128, CCH, 0, 0);
  zero_kernel<<<PLB, 256, 0, stream>>>(Mp, PLN);
  emb_scatter_kernel<<<ECB, 256, 0, stream>>>(ei, ev, w1h, w2h, w3h, zij, Mp);
  tn_ln64_kernel<<<NN, 64, 0, stream>>>(Mp, F(14), F(13), normH);
  f2h_kernel<<<PLB, 256, 0, stream>>>(Mp, Chh, PLN);
  gemm(Chh, elt[0], nullptr, nullptr, nullptr, Yp, NN, CCH, CCH, 0, 0);
  for (int pl = 1; pl <= 3; ++pl)
    gemm(Chh + pl * NCF, elt[1], nullptr, nullptr, nullptr, Yp + pl * NCF, NN, CCH, CCH, 0, 0);
  for (int pl = 4; pl <= 9; ++pl)
    gemm(Chh + pl * NCF, elt[2], nullptr, nullptr, nullptr, Yp + pl * NCF, NN, CCH, CCH, 0, 0);
  gemm(normH, ls1w, F(15), nullptr, ls1o, nullptr, NN, CCH, 128, 1, 0);
  gemm(ls1o,  ls2w, F(17), nullptr, ls2o, nullptr, NN, 128, 192, 1, 0);
  emb_combine_kernel<<<NCB, 256, 0, stream>>>(Yp, ls2o, Xp);

  // ---------- interaction layers ----------
  for (int l = 0; l < 2; ++l) {
    gemm(eattr, l1w[l], F(pb[l] + 0), nullptr, ea1, nullptr, EE, RR, CCH, 1, 0);
    gemm(ea1,   l2w[l], F(pb[l] + 2), nullptr, ea2, nullptr, EE, CCH, 128, 1, 0);
    gemm(ea2,   l3w[l], F(pb[l] + 4), ccv,     ea3, nullptr, EE, 128, 192, 1, 1);
    x_normalize_kernel<<<NCB, 256, 0, stream>>>(Xp, Xh);
    gemm(Xh, llt[l][0], nullptr, nullptr, nullptr, Yp, NN, CCH, CCH, 0, 0);
    for (int pl = 1; pl <= 3; ++pl)
      gemm(Xh + pl * NCF, llt[l][1], nullptr, nullptr, nullptr, Yp + pl * NCF, NN, CCH, CCH, 0, 0);
    for (int pl = 4; pl <= 9; ++pl)
      gemm(Xh + pl * NCF, llt[l][2], nullptr, nullptr, nullptr, Yp + pl * NCF, NN, CCH, CCH, 0, 0);
    zero_kernel<<<PLB, 256, 0, stream>>>(Mp, PLN);
    inter_scatter_kernel<<<ECB, 256, 0, stream>>>(ei, ea3, Yp, Mp);
    inter_combine_kernel<<<NCB, 256, 0, stream>>>(Mp, Yp, Chh);
    gemm(Chh, llt[l][3], nullptr, nullptr, nullptr, Dp, NN, CCH, CCH, 0, 0);
    for (int pl = 1; pl <= 3; ++pl)
      gemm(Chh + pl * NCF, llt[l][4], nullptr, nullptr, nullptr, Dp + pl * NCF, NN, CCH, CCH, 0, 0);
    for (int pl = 4; pl <= 9; ++pl)
      gemm(Chh + pl * NCF, llt[l][5], nullptr, nullptr, nullptr, Dp + pl * NCF, NN, CCH, CCH, 0, 0);
    x_update_kernel<<<NCB, 256, 0, stream>>>(Dp, Xp);
  }

  // ---------- output head ----------
  final_ln_kernel<<<NN, 256, 0, stream>>>(Xp, F(51), F(50), xcat);
  gemm(xcat, outw, F(48), nullptr, nullptr, (float*)d_out, NN, 192, CCH, 1, 0);
}
